// ZernikeRankPooling_84877143704110
// MI455X (gfx1250) — compile-verified, matrix operand order checked by reading, not run on hardware
//
#include <hip/hip_runtime.h>
#include <hip/hip_bf16.h>
#include <math.h>

typedef _Float16 v8h  __attribute__((ext_vector_type(8)));
typedef _Float16 v16h __attribute__((ext_vector_type(16)));
typedef float    v8f  __attribute__((ext_vector_type(8)));

// ---------------------------------------------------------------------------
// Kernel 1: Zernike pooling.
// One wave32 per (batch, rank) image; lane = channel (32 channels per rank).
// x layout: (B=1024, H=16, W=16, C=96) -> per pixel the 96 channels are
// contiguous, so a wave's per-pixel load is one contiguous 128B segment.
// Pass A: m00 / centroid.  Pass B: masked polynomial moments (no trig needed).
// Writes pooled (f32, for BN stats) and pooled_h (f16, for the WMMA head).
// ---------------------------------------------------------------------------
__global__ __launch_bounds__(256) void zernike_pool_kernel(
    const float* __restrict__ x,
    float* __restrict__ pooled,
    _Float16* __restrict__ pooled_h)
{
    const int lane = threadIdx.x & 31;
    const int img  = blockIdx.x * 8 + (threadIdx.x >> 5);   // 0..3071
    const int b    = img / 3;
    const int rank = img - b * 3;
    const float* base = x + (size_t)b * 256 * 96 + rank * 32 + lane;

    const float step = 2.0f / 15.0f;

    // ---- pass A: raw moments for center of mass ----
    float m00 = 0.0f, sx = 0.0f, sy = 0.0f;
#pragma unroll 4
    for (int p = 0; p < 256; ++p) {
        float v = base[(size_t)p * 96];
        float X = (float)(p & 15) * step - 1.0f;
        float Y = (float)(p >> 4) * step - 1.0f;
        m00 += v;
        sx  += v * X;
        sy  += v * Y;
    }
    float cx = sx / (m00 + 1e-6f);
    float cy = sy / (m00 + 1e-6f);

    // ---- pass B: masked Zernike sums (polynomial forms, no trig) ----
    float t0 = 0.f, t1x = 0.f, t1y = 0.f, t2 = 0.f, t2x = 0.f, t2y = 0.f;
    float t3x = 0.f, t3y = 0.f, t4x = 0.f, t4y = 0.f;
#pragma unroll 2
    for (int p = 0; p < 256; ++p) {
        float v  = base[(size_t)p * 96];
        float Xs = (float)(p & 15) * step - 1.0f - cx;
        float Ys = (float)(p >> 4) * step - 1.0f - cy;
        float xx = Xs * Xs, yy = Ys * Ys, xy = Xs * Ys;
        float r2 = xx + yy + 1e-12f;
        float fm = (r2 <= 1.0f) ? v : 0.0f;

        t0  += fm;                              // (0,0): R=1
        t1x += fm * Xs;                         // (1,1): rho*cos = Xs
        t1y += fm * Ys;                         //        rho*sin = Ys
        t2  += fm * (2.0f * r2 - 1.0f);         // (2,0)
        t2x += fm * (xx - yy);                  // (2,2): r2*cos2t
        t2y += fm * (2.0f * xy);                //        r2*sin2t
        float w3 = 3.0f * r2 - 2.0f;            // (3,1)
        t3x += fm * w3 * Xs;
        t3y += fm * w3 * Ys;
        t4x += fm * (4.0f * Xs * xx - 3.0f * r2 * Xs);  // (3,3): rho^3*cos3t
        t4y += fm * (3.0f * r2 * Ys - 4.0f * Ys * yy);  //        rho^3*sin3t
    }

    const float invm  = 1.0f / 256.0f;
    const float invpi = 0.31830988618379067f;
    float r0  = t0  * invm;
    float r1x = t1x * invm, r1y = t1y * invm;
    float r2m = t2  * invm;
    float r2x = t2x * invm, r2y = t2y * invm;
    float r3x = t3x * invm, r3y = t3y * invm;
    float r4x = t4x * invm, r4y = t4y * invm;

    float o0 = 1.0f * invpi * sqrtf(r0  * r0  + 1e-12f);
    float o1 = 2.0f * invpi * sqrtf(r1x * r1x + r1y * r1y + 1e-12f);
    float o2 = 3.0f * invpi * sqrtf(r2m * r2m + 1e-12f);
    float o3 = 3.0f * invpi * sqrtf(r2x * r2x + r2y * r2y + 1e-12f);
    float o4 = 4.0f * invpi * sqrtf(r3x * r3x + r3y * r3y + 1e-12f);
    float o5 = 4.0f * invpi * sqrtf(r4x * r4x + r4y * r4y + 1e-12f);

    // feature index = rank*192 + inv*32 + channel
    size_t fbase = (size_t)b * 576 + rank * 192 + lane;
    float*    op = pooled   + fbase;
    _Float16* oh = pooled_h + fbase;
    op[0 * 32] = o0;  oh[0 * 32] = (_Float16)o0;
    op[1 * 32] = o1;  oh[1 * 32] = (_Float16)o1;
    op[2 * 32] = o2;  oh[2 * 32] = (_Float16)o2;
    op[3 * 32] = o3;  oh[3 * 32] = (_Float16)o3;
    op[4 * 32] = o4;  oh[4 * 32] = (_Float16)o4;
    op[5 * 32] = o5;  oh[5 * 32] = (_Float16)o5;
}

// ---------------------------------------------------------------------------
// Kernel 2: BatchNorm batch stats. One thread per feature (coalesced column
// reads: consecutive lanes = consecutive features). Biased variance.
// ---------------------------------------------------------------------------
__global__ __launch_bounds__(32) void bn_stats_kernel(
    const float* __restrict__ pooled,
    float* __restrict__ mu,
    float* __restrict__ rsig)
{
    int f = blockIdx.x * 32 + threadIdx.x;   // 0..575 (grid 18 x 32)
    float s = 0.0f, s2 = 0.0f;
    for (int row = 0; row < 1024; ++row) {
        float v = pooled[(size_t)row * 576 + f];
        s  += v;
        s2 += v * v;
    }
    float m   = s * (1.0f / 1024.0f);
    float var = s2 * (1.0f / 1024.0f) - m * m;
    mu[f]   = m;
    rsig[f] = rsqrtf(var + 1e-5f);
}

// ---------------------------------------------------------------------------
// Kernel 3: fold BN affine into the head weights.
//   W_eff[o,f] = W[o,f] * gamma[f] * rsig[f]           (f16, padded to 16 rows)
//   b_eff[o]   = b[o] + sum_f W[o,f]*(beta[f] - mu[f]*gamma[f]*rsig[f])
// grid = 16 (output rows, >=10 zero-padded), block = 576 (one thread/feature).
// ---------------------------------------------------------------------------
__global__ __launch_bounds__(576) void fold_bn_kernel(
    const float* __restrict__ W_lin, const float* __restrict__ b_lin,
    const float* __restrict__ gamma, const float* __restrict__ beta,
    const float* __restrict__ mu,    const float* __restrict__ rsig,
    _Float16* __restrict__ Weff,     float* __restrict__ bias_eff)
{
    const int o = blockIdx.x;      // 0..15
    const int f = threadIdx.x;     // 0..575
    __shared__ float ssum;
    if (f == 0) ssum = 0.0f;
    __syncthreads();
    if (o < 10) {
        float g = gamma[f] * rsig[f];
        float w = W_lin[o * 576 + f];
        Weff[(size_t)o * 576 + f] = (_Float16)(w * g);
        atomicAdd(&ssum, w * (beta[f] - mu[f] * g));
    } else {
        Weff[(size_t)o * 576 + f] = (_Float16)0.0f;
    }
    __syncthreads();
    if (f == 0) bias_eff[o] = (o < 10) ? (b_lin[o] + ssum) : 0.0f;
}

// ---------------------------------------------------------------------------
// Kernel 4: head GEMM via v_wmma_f32_16x16x32_f16.
// out(1024x10) = pooled_h(1024x576) @ Weff^T(576x16)[:, :10] + b_eff
// One wave per 16-row tile (64 tiles); K = 576 = 18 x 32.
// A frag (16x32 f16): lane m=lane%16, halves [0..7]=K(8g..8g+7),
//                     [8..15]=K(16+8g..16+8g+7), g=lane/16.
// B frag (32x16 f16): same K pattern with n=lane%16 (B[k][n]=Weff[n][k]).
// C/D (16x16 f32):    elem i -> M = i + 8g, N = lane%16.
// ---------------------------------------------------------------------------
__global__ __launch_bounds__(32) void head_wmma_kernel(
    const _Float16* __restrict__ pooled_h,
    const _Float16* __restrict__ Weff,
    const float* __restrict__ bias_eff,
    float* __restrict__ out)
{
    const int lane = threadIdx.x & 31;
    const int g = lane >> 4;
    const int r = lane & 15;
    const int m0 = blockIdx.x * 16;

    const _Float16* Arow = pooled_h + (size_t)(m0 + r) * 576;
    const _Float16* Brow = Weff     + (size_t)r * 576;

    v8f acc = {};
#pragma unroll 3
    for (int kt = 0; kt < 18; ++kt) {
        int k0 = kt * 32 + g * 8;
        v8h alo = *(const v8h*)(Arow + k0);
        v8h ahi = *(const v8h*)(Arow + k0 + 16);
        v8h blo = *(const v8h*)(Brow + k0);
        v8h bhi = *(const v8h*)(Brow + k0 + 16);
        v16h a  = __builtin_shufflevector(alo, ahi, 0,1,2,3,4,5,6,7,8,9,10,11,12,13,14,15);
        v16h bb = __builtin_shufflevector(blo, bhi, 0,1,2,3,4,5,6,7,8,9,10,11,12,13,14,15);
        acc = __builtin_amdgcn_wmma_f32_16x16x32_f16(
            /*neg_a=*/false, a, /*neg_b=*/false, bb,
            /*c_mod=*/(short)0, acc, /*reuse_a=*/false, /*reuse_b=*/false);
    }

    if (r < 10) {
        float bias = bias_eff[r];
#pragma unroll
        for (int i = 0; i < 8; ++i) {
            int m = m0 + i + 8 * g;
            out[(size_t)m * 10 + r] = acc[i] + bias;
        }
    }
}

// ---------------------------------------------------------------------------
// Workspace layout (bytes, all 256B-aligned):
//   pooled   f32 1024x576 : [0,          2359296)
//   pooled_h f16 1024x576 : [2359296,    3538944)
//   mu       f32 576      : [3538944,    3541248)
//   rsig     f32 576      : [3541248,    3543552)
//   Weff     f16 16x576   : [3543552,    3561984)
//   bias_eff f32 16       : [3561984,    3562048)
// ---------------------------------------------------------------------------
extern "C" void kernel_launch(void* const* d_in, const int* in_sizes, int n_in,
                              void* d_out, int out_size, void* d_ws, size_t ws_size,
                              hipStream_t stream) {
    (void)in_sizes; (void)n_in; (void)out_size; (void)ws_size;

    const float* x     = (const float*)d_in[0];
    const float* gamma = (const float*)d_in[1];
    const float* beta  = (const float*)d_in[2];
    const float* W_lin = (const float*)d_in[3];
    const float* b_lin = (const float*)d_in[4];
    float* out = (float*)d_out;

    char* ws = (char*)d_ws;
    float*    pooled   = (float*)(ws + 0);
    _Float16* pooled_h = (_Float16*)(ws + 2359296);
    float*    mu       = (float*)(ws + 3538944);
    float*    rsig     = (float*)(ws + 3541248);
    _Float16* Weff     = (_Float16*)(ws + 3543552);
    float*    bias_eff = (float*)(ws + 3561984);

    // 3072 (batch, rank) images, 8 waves (one image each) per 256-thread block
    zernike_pool_kernel<<<384, 256, 0, stream>>>(x, pooled, pooled_h);

    // 576 features, one thread each
    bn_stats_kernel<<<18, 32, 0, stream>>>(pooled, mu, rsig);

    // fold BN into head weights (16 padded output rows)
    fold_bn_kernel<<<16, 576, 0, stream>>>(W_lin, b_lin, gamma, beta,
                                           mu, rsig, Weff, bias_eff);

    // WMMA head GEMM: 64 M-tiles of 16 rows, one wave each
    head_wmma_kernel<<<64, 32, 0, stream>>>(pooled_h, Weff, bias_eff, out);
}
